// ColbertLoss_27350351741579
// MI455X (gfx1250) — matrix operand order repaired and checked by reading, not compiled, not run on hardware
//
#include <hip/hip_runtime.h>
#include <hip/hip_bf16.h>
#include <stdint.h>

typedef __attribute__((ext_vector_type(16))) _Float16 v16h;
typedef __attribute__((ext_vector_type(8)))  float    v8f;
typedef __attribute__((ext_vector_type(4)))  uint32_t v4u;

union frag16 { v16h h; uint32_t u[8]; v4u q[2]; };

#define QTOK 32
#define DTOK 1024
#define DIM  128
#define NB   64
#define NC   64
#define TEMPERATURE 0.02f

// LDS tile: 16 doc tokens x 128 dims of f16, row padded 256B -> 272B (136 halfs)
#define ROWH 136   // halfs per padded row
#define NT   (DTOK / 16)   // 64 doc-token tiles

// ---------------------------------------------------------------------------
// Kernel 1: f32 -> f16 conversion (4-wide)
// ---------------------------------------------------------------------------
struct h4 { _Float16 x, y, z, w; };

__global__ void cvt_f32_to_f16_kernel(const float* __restrict__ src,
                                      _Float16* __restrict__ dst, int n4) {
  int i = blockIdx.x * blockDim.x + threadIdx.x;
  if (i < n4) {
    float4 v = reinterpret_cast<const float4*>(src)[i];
    h4 o;
    o.x = (_Float16)v.x; o.y = (_Float16)v.y;
    o.z = (_Float16)v.z; o.w = (_Float16)v.w;
    reinterpret_cast<h4*>(dst)[i] = o;
  }
}

// ---------------------------------------------------------------------------
// Kernel 2: MaxSim scores via v_wmma_f32_16x16x32_f16 with async-LDS staging.
// grid = (64 c, 16 b-groups), block = 128 threads (4 waves).
// Wave w owns b = bg*4 + w; all 4 waves share the d[c] tile staged in LDS by
// GLOBAL_LOAD_ASYNC_TO_LDS_B128 (double-buffered, ASYNCcnt-tracked).
// S[b*NC + c] = sum_n max_s ( q[b,n,:] . d[c,s,:] )   (unnormalized)
// ---------------------------------------------------------------------------
__global__ void __launch_bounds__(128)
maxsim_kernel(const _Float16* __restrict__ q16,
              const _Float16* __restrict__ d16,
              float* __restrict__ S) {
  const int c    = blockIdx.x;
  const int bg   = blockIdx.y;
  const int wave = threadIdx.x >> 5;   // 0..3 -> which b this wave owns
  const int lane = threadIdx.x & 31;
  const int half = lane >> 4;          // 0 or 1
  const int l15  = lane & 15;
  const int b    = bg * 4 + wave;

  const _Float16* qb = q16 + (size_t)b * QTOK * DIM;
  const _Float16* dc = d16 + (size_t)c * DTOK * DIM;

  // Double-buffered doc-token tile: [2][16 rows][136 halfs] (272B padded rows)
  __shared__ __align__(16) _Float16 tile[2][16 * ROWH];

  // Each of the 128 threads async-copies one 16B chunk per tile:
  // thread t -> row = t>>3 (doc token in tile), col = t&7 (16B chunk of dim)
  const int arow = threadIdx.x >> 3;
  const int acol = threadIdx.x & 7;

  auto issue_tile = [&](int nt, int bufi) {
    const _Float16* g = dc + (size_t)(nt * 16 + arow) * DIM + acol * 8;
    uint32_t laddr =
        (uint32_t)(uintptr_t)&tile[bufi][arow * ROWH + acol * 8];
    uint64_t gaddr = (uint64_t)(uintptr_t)g;
    asm volatile("global_load_async_to_lds_b128 %0, %1, off"
                 :: "v"(laddr), "v"(gaddr)
                 : "memory");
  };

  // --- Preload this wave's q[b] as A fragments: 2 M-tiles x 4 K-steps -----
  // 16-bit A 16x32 layout: lane holds row m = l15 (+16*mt);
  // VGPR i holds K = (i<4 ? 2i : 2i+8) + 8*half  (+32*ks), pairs contiguous.
  frag16 A[2][4];
#pragma unroll
  for (int mt = 0; mt < 2; ++mt) {
    const int m = l15 + 16 * mt;
#pragma unroll
    for (int ks = 0; ks < 4; ++ks) {
#pragma unroll
      for (int i = 0; i < 8; ++i) {
        const int k = ks * 32 + ((i < 4) ? 2 * i : 2 * i + 8) + 8 * half;
        A[mt][ks].u[i] = *(const uint32_t*)(qb + m * DIM + k);  // 2 f16
      }
    }
  }

  // Prologue: stage tile 0
  issue_tile(0, 0);
  asm volatile("s_wait_asynccnt 0x0" ::: "memory");
  __syncthreads();

  // Running row-max in C/D layout: lane half 0 -> m = r, half 1 -> m = 8+r
  float rowmax[2][8];
#pragma unroll
  for (int mt = 0; mt < 2; ++mt)
#pragma unroll
    for (int r = 0; r < 8; ++r) rowmax[mt][r] = -1e30f;

  for (int nt = 0; nt < NT; ++nt) {
    const int cur = nt & 1;
    if (nt + 1 < NT) issue_tile(nt + 1, (nt + 1) & 1);

    // 16-bit B 32x16 layout: lane holds column n = l15 (doc token in tile);
    // elements 0-7 hold K=8*half+0..7, elements 8-15 hold K=16+8*half+0..7.
    // Per lane, per K-step: two 16-byte chunks from the LDS row.
    frag16 B[4];
    const _Float16* brow = &tile[cur][l15 * ROWH + 8 * half];
#pragma unroll
    for (int ks = 0; ks < 4; ++ks) {
      B[ks].q[0] = *(const v4u*)(brow + ks * 32);
      B[ks].q[1] = *(const v4u*)(brow + ks * 32 + 16);
    }

#pragma unroll
    for (int mt = 0; mt < 2; ++mt) {
      v8f acc = {0.f, 0.f, 0.f, 0.f, 0.f, 0.f, 0.f, 0.f};
#pragma unroll
      for (int ks = 0; ks < 4; ++ks) {
        acc = __builtin_amdgcn_wmma_f32_16x16x32_f16(
            /*neg_a=*/false, A[mt][ks].h,
            /*neg_b=*/false, B[ks].h,
            /*c_mod=*/(short)0, acc,
            /*reuse_a=*/false, /*reuse_b=*/false);
      }
#pragma unroll
      for (int r = 0; r < 8; ++r)
        rowmax[mt][r] = fmaxf(rowmax[mt][r], acc[r]);
    }

    // Next tile landed (own asyncs), then all waves done reading `cur`
    asm volatile("s_wait_asynccnt 0x0" ::: "memory");
    __syncthreads();
  }

  // --- Reduce max over n within each 16-lane half-group (xor 1,2,4,8) -----
#pragma unroll
  for (int mt = 0; mt < 2; ++mt)
#pragma unroll
    for (int r = 0; r < 8; ++r) {
      float v = rowmax[mt][r];
#pragma unroll
      for (int off = 8; off >= 1; off >>= 1)
        v = fmaxf(v, __shfl_xor(v, off, 32));
      rowmax[mt][r] = v;
    }

  // Each lane now holds the final row-max for its 16 m values
  // (m = mt*16 + half*8 + r); sum them, then add across halves.
  float psum = 0.f;
#pragma unroll
  for (int mt = 0; mt < 2; ++mt)
#pragma unroll
    for (int r = 0; r < 8; ++r) psum += rowmax[mt][r];
  psum += __shfl_xor(psum, 16, 32);

  if (lane == 0) S[b * NC + c] = psum;
}

// ---------------------------------------------------------------------------
// Kernel 3: normalize, log-softmax over c, NLL at label b+offset, mean.
// 1 block, 64 threads (thread b owns row b). Two streaming passes over S
// (no dynamically-indexed private array -> no scratch).
// ---------------------------------------------------------------------------
__global__ void __launch_bounds__(NB)
loss_kernel(const float* __restrict__ S,
            const float* __restrict__ qf32,
            const int* __restrict__ offset_p,
            float* __restrict__ out) {
  __shared__ float partial[NB];
  const int b = threadIdx.x;

  // length = count of non-padding query tokens (q[b,n,0] != 0)
  float len = 0.f;
  for (int n = 0; n < QTOK; ++n)
    len += (qf32[(size_t)b * QTOK * DIM + (size_t)n * DIM] != 0.f) ? 1.f : 0.f;
  const float inv = 1.0f / (len * TEMPERATURE);

  const int label = b + offset_p[0];

  float mx = -1e30f;
  for (int c2 = 0; c2 < NC; ++c2)
    mx = fmaxf(mx, S[b * NC + c2] * inv);

  float sum = 0.f;
  float lab = 0.f;
  for (int c2 = 0; c2 < NC; ++c2) {
    const float lg = S[b * NC + c2] * inv;
    sum += expf(lg - mx);
    if (c2 == label) lab = lg;
  }

  partial[b] = -((lab - mx) - logf(sum));
  __syncthreads();

  if (b == 0) {
    float acc = 0.f;
    for (int i = 0; i < NB; ++i) acc += partial[i];
    out[0] = acc / (float)NB;
  }
}

// ---------------------------------------------------------------------------
extern "C" void kernel_launch(void* const* d_in, const int* in_sizes, int n_in,
                              void* d_out, int out_size, void* d_ws, size_t ws_size,
                              hipStream_t stream) {
  const float* q   = (const float*)d_in[0];   // (64, 32, 128) f32
  const float* dd  = (const float*)d_in[1];   // (64, 1024, 128) f32
  const int*  offs = (const int*)d_in[2];     // scalar offset
  float* out = (float*)d_out;

  // Workspace layout: q16 (512 KB) | d16 (16 MB) | S (16 KB)
  const size_t nq = (size_t)NB * QTOK * DIM;   //   262144
  const size_t nd = (size_t)NC * DTOK * DIM;   //  8388608
  _Float16* q16 = (_Float16*)d_ws;
  _Float16* d16 = q16 + nq;
  float*    S   = (float*)(d16 + nd);

  {
    const int n4 = (int)(nq / 4);
    cvt_f32_to_f16_kernel<<<(n4 + 255) / 256, 256, 0, stream>>>(q, q16, n4);
  }
  {
    const int n4 = (int)(nd / 4);
    cvt_f32_to_f16_kernel<<<(n4 + 255) / 256, 256, 0, stream>>>(dd, d16, n4);
  }

  dim3 grid(NC, NB / 4);
  maxsim_kernel<<<grid, 128, 0, stream>>>(q16, d16, S);

  loss_kernel<<<1, NB, 0, stream>>>(S, q, offs, out);
}
